// LSTMAutoencoderModel_67997922230927
// MI455X (gfx1250) — matrix operand order repaired and checked by reading, not compiled, not run on hardware
//
#include <hip/hip_runtime.h>
#include <hip/hip_bf16.h>

#define B_ 512
#define S_ 512
#define I_ 32
#define H_ 128
#define G_ 512   // 4*H

typedef __attribute__((ext_vector_type(16))) _Float16 v16h;
typedef __attribute__((ext_vector_type(8)))  _Float16 v8h;
typedef __attribute__((ext_vector_type(8)))  float    v8f;

// ---------------------------------------------------------------------------
// WMMA fragment loaders (CDNA5 ISA 7.12.2, 16-bit layouts, wave32)
// A 16x32: lane m=l&15; halves 0..7 hold K = (l>=16?8:0)+0..7, halves 8..15
//          hold K = 16+(l>=16?8:0)+0..7  -> two contiguous 16B loads.
// B 32x16: lane n=l&15; same K order as A (halves 0..7 K=(l>=16?8:0)+0..7,
//          halves 8..15 K=16+(l>=16?8:0)+0..7) -> two contiguous 16B loads.
// Both come straight from row-major [rows x K] f16 arrays.
// ---------------------------------------------------------------------------
__device__ __forceinline__ v16h ld_a(const _Float16* __restrict__ row, int kb) {
  v8h lo = *(const v8h*)(row + kb);
  v8h hi = *(const v8h*)(row + kb + 16);
  v16h r;
#pragma unroll
  for (int i = 0; i < 8; ++i) { r[i] = lo[i]; r[i + 8] = hi[i]; }
  return r;
}
__device__ __forceinline__ v16h ld_b(const _Float16* __restrict__ row, int kc, int lhi) {
  return ld_a(row, kc + lhi * 8);
}
__device__ __forceinline__ v8f wmma16(v16h a, v16h b, v8f c) {
  return __builtin_amdgcn_wmma_f32_16x16x32_f16(false, a, false, b, (short)0, c,
                                                false, false);
}
__device__ __forceinline__ float sigm(float x) { return 1.0f / (1.0f + __expf(-x)); }

// ---------------------------------------------------------------------------
// Encoder step: gates = x_t @ Wih.T + h @ Whh.T + bsum ; update (h,c).
// grid = (B/16, H/16), block = 32 (one wave); each wave owns a 16x16
// (batch x hidden) tile and computes all 4 gate tiles for it.
// ---------------------------------------------------------------------------
__global__ void __launch_bounds__(32)
enc_step(const _Float16* __restrict__ xh, int t,
         const _Float16* __restrict__ Wih,   // [4H x I]
         const _Float16* __restrict__ Whh,   // [4H x H]
         const float*    __restrict__ bsum,  // [4H]
         const _Float16* __restrict__ h_in,  // [B x H]
         _Float16*       __restrict__ h_out, // [B x H]
         float*          __restrict__ c)     // [B x H]
{
  const int lane  = threadIdx.x;
  const int nl    = lane & 15;
  const int lhi   = lane >> 4;
  const int khalf = lhi * 8;
  const int mbase = blockIdx.x * 16;
  const int jbase = blockIdx.y * 16;

  const v8f z8 = {0.f, 0.f, 0.f, 0.f, 0.f, 0.f, 0.f, 0.f};
  v8f acc[4] = {z8, z8, z8, z8};

  // input projection, K = I = 32 (one k-chunk)
  {
    const _Float16* arow = xh + ((size_t)(mbase + nl) * S_ + t) * I_;
    v16h a = ld_a(arow, khalf);
#pragma unroll
    for (int g = 0; g < 4; ++g) {
      v16h b = ld_b(Wih + (size_t)(g * H_ + jbase + nl) * I_, 0, lhi);
      acc[g] = wmma16(a, b, acc[g]);
    }
  }
  // recurrent projection, K = H = 128
  const _Float16* arow = h_in + (size_t)(mbase + nl) * H_;
#pragma unroll
  for (int kc = 0; kc < H_; kc += 32) {
    v16h a = ld_a(arow, kc + khalf);
#pragma unroll
    for (int g = 0; g < 4; ++g) {
      v16h b = ld_b(Whh + (size_t)(g * H_ + jbase + nl) * H_, kc, lhi);
      acc[g] = wmma16(a, b, acc[g]);
    }
  }

  const int n = jbase + nl;
  const float bI = bsum[0 * H_ + n], bF = bsum[1 * H_ + n];
  const float bG = bsum[2 * H_ + n], bO = bsum[3 * H_ + n];
  const int mh = lhi * 8;  // C/D layout: vgpr r -> M = r + (lane>=16 ? 8 : 0)
#pragma unroll
  for (int r = 0; r < 8; ++r) {
    const int m = mbase + mh + r;
    const size_t idx = (size_t)m * H_ + n;
    float gi = sigm(acc[0][r] + bI);
    float gf = sigm(acc[1][r] + bF);
    float gg = tanhf(acc[2][r] + bG);
    float go = sigm(acc[3][r] + bO);
    float cn = gf * c[idx] + gi * gg;
    c[idx] = cn;
    h_out[idx] = (_Float16)(go * tanhf(cn));
  }
}

// fc tile helper: frame = h @ fcW.T + fcb, written to out[:, t, :]
__device__ __forceinline__ void fc_tile(const _Float16* __restrict__ hsrc,
                                        const _Float16* __restrict__ fcW,  // [I x H]
                                        const float* __restrict__ fcb,
                                        float* __restrict__ out, int t,
                                        int mbase, int nb, int lane) {
  const int nl = lane & 15, lhi = lane >> 4, khalf = lhi * 8;
  v8f acc = {0.f, 0.f, 0.f, 0.f, 0.f, 0.f, 0.f, 0.f};
  const _Float16* arow = hsrc + (size_t)(mbase + nl) * H_;
#pragma unroll
  for (int kc = 0; kc < H_; kc += 32) {
    v16h a = ld_a(arow, kc + khalf);
    v16h b = ld_b(fcW + (size_t)(nb + nl) * H_, kc, lhi);
    acc = wmma16(a, b, acc);
  }
  const int n = nb + nl;
  const float bb = fcb[n];
  const int mh = lhi * 8;
#pragma unroll
  for (int r = 0; r < 8; ++r) {
    const int m = mbase + mh + r;
    out[(size_t)m * (S_ * I_) + (size_t)t * I_ + n] = acc[r] + bb;
  }
}

// ---------------------------------------------------------------------------
// Decoder step t: gates = A_in @ W.T + bsum  (W = Whh for t==0, Wih+Whh after,
// since inp == h for t>=1); cell state is enc_c every step (not persisted).
// grid = (B/16, 10): y<8 -> cell tiles, y in {8,9} -> fc for frame t-1.
// ---------------------------------------------------------------------------
__global__ void __launch_bounds__(32)
dec_step(const _Float16* __restrict__ A_in,
         const _Float16* __restrict__ W,      // [4H x H]
         const float*    __restrict__ bsum,   // [4H]
         const float*    __restrict__ enc_c,  // [B x H]
         _Float16*       __restrict__ h_out,  // [B x H]
         const _Float16* __restrict__ fc_src, // h of step t-1
         const _Float16* __restrict__ fcW,    // [I x H]
         const float*    __restrict__ fcb,
         float*          __restrict__ out, int fc_t)
{
  const int lane  = threadIdx.x;
  const int nl    = lane & 15;
  const int lhi   = lane >> 4;
  const int khalf = lhi * 8;
  const int mbase = blockIdx.x * 16;

  if (blockIdx.y < 8) {
    const int jbase = blockIdx.y * 16;
    const v8f z8 = {0.f, 0.f, 0.f, 0.f, 0.f, 0.f, 0.f, 0.f};
    v8f acc[4] = {z8, z8, z8, z8};
    const _Float16* arow = A_in + (size_t)(mbase + nl) * H_;
#pragma unroll
    for (int kc = 0; kc < H_; kc += 32) {
      v16h a = ld_a(arow, kc + khalf);
#pragma unroll
      for (int g = 0; g < 4; ++g) {
        v16h b = ld_b(W + (size_t)(g * H_ + jbase + nl) * H_, kc, lhi);
        acc[g] = wmma16(a, b, acc[g]);
      }
    }
    const int n = jbase + nl;
    const float bI = bsum[0 * H_ + n], bF = bsum[1 * H_ + n];
    const float bG = bsum[2 * H_ + n], bO = bsum[3 * H_ + n];
    const int mh = lhi * 8;
#pragma unroll
    for (int r = 0; r < 8; ++r) {
      const int m = mbase + mh + r;
      const size_t idx = (size_t)m * H_ + n;
      float gi = sigm(acc[0][r] + bI);
      float gf = sigm(acc[1][r] + bF);
      float gg = tanhf(acc[2][r] + bG);
      float go = sigm(acc[3][r] + bO);
      float cn = gf * enc_c[idx] + gi * gg;     // decoder c is discarded
      h_out[idx] = (_Float16)(go * tanhf(cn));
    }
  } else {
    if (fc_t < 0) return;
    fc_tile(fc_src, fcW, fcb, out, fc_t, mbase, (blockIdx.y - 8) * 16, lane);
  }
}

__global__ void __launch_bounds__(32)
fc_out_kernel(const _Float16* __restrict__ hsrc, const _Float16* __restrict__ fcW,
              const float* __restrict__ fcb, float* __restrict__ out, int t) {
  fc_tile(hsrc, fcW, fcb, out, t, blockIdx.x * 16, blockIdx.y * 16, threadIdx.x);
}

// ------------------------------ prep kernels -------------------------------
__global__ void cvt_f16(const float* __restrict__ s, _Float16* __restrict__ d, int n) {
  int i = blockIdx.x * blockDim.x + threadIdx.x;
  if (i < n) d[i] = (_Float16)s[i];
}
__global__ void addcvt_f16(const float* __restrict__ a, const float* __restrict__ b,
                           _Float16* __restrict__ d, int n) {
  int i = blockIdx.x * blockDim.x + threadIdx.x;
  if (i < n) d[i] = (_Float16)(a[i] + b[i]);
}
__global__ void add_f32(const float* __restrict__ a, const float* __restrict__ b,
                        float* __restrict__ d, int n) {
  int i = blockIdx.x * blockDim.x + threadIdx.x;
  if (i < n) d[i] = a[i] + b[i];
}
__global__ void zero16(_Float16* __restrict__ d, int n) {
  int i = blockIdx.x * blockDim.x + threadIdx.x;
  if (i < n) d[i] = (_Float16)0.f;
}
__global__ void zero32(float* __restrict__ d, int n) {
  int i = blockIdx.x * blockDim.x + threadIdx.x;
  if (i < n) d[i] = 0.f;
}

// ---------------------------------------------------------------------------
extern "C" void kernel_launch(void* const* d_in, const int* in_sizes, int n_in,
                              void* d_out, int out_size, void* d_ws, size_t ws_size,
                              hipStream_t stream) {
  (void)in_sizes; (void)n_in; (void)out_size; (void)ws_size;
  const float* x    = (const float*)d_in[0];
  const float* eWih = (const float*)d_in[1];
  const float* eWhh = (const float*)d_in[2];
  const float* ebih = (const float*)d_in[3];
  const float* ebhh = (const float*)d_in[4];
  const float* dWih = (const float*)d_in[5];
  const float* dWhh = (const float*)d_in[6];
  const float* dbih = (const float*)d_in[7];
  const float* dbhh = (const float*)d_in[8];
  const float* fcW  = (const float*)d_in[9];
  const float* fcb  = (const float*)d_in[10];
  float* out = (float*)d_out;

  char* ws = (char*)d_ws;
  size_t off = 0;
  auto alloc = [&](size_t bytes) -> void* {
    off = (off + 255) & ~(size_t)255;
    void* p = ws + off;
    off += bytes;
    return p;
  };
  _Float16* xh   = (_Float16*)alloc((size_t)B_ * S_ * I_ * 2);
  _Float16* wihE = (_Float16*)alloc((size_t)G_ * I_ * 2);
  _Float16* whhE = (_Float16*)alloc((size_t)G_ * H_ * 2);
  _Float16* whhD = (_Float16*)alloc((size_t)G_ * H_ * 2);
  _Float16* wsum = (_Float16*)alloc((size_t)G_ * H_ * 2);
  _Float16* fcWh = (_Float16*)alloc((size_t)I_ * H_ * 2);
  float*    bsE  = (float*)alloc((size_t)G_ * 4);
  float*    bsD  = (float*)alloc((size_t)G_ * 4);
  _Float16* hE0  = (_Float16*)alloc((size_t)B_ * H_ * 2);
  _Float16* hE1  = (_Float16*)alloc((size_t)B_ * H_ * 2);
  float*    cE   = (float*)alloc((size_t)B_ * H_ * 4);
  _Float16* hD0  = (_Float16*)alloc((size_t)B_ * H_ * 2);
  _Float16* hD1  = (_Float16*)alloc((size_t)B_ * H_ * 2);

  auto nb = [](int n) { return (n + 255) / 256; };
  cvt_f16<<<nb(B_ * S_ * I_), 256, 0, stream>>>(x, xh, B_ * S_ * I_);
  cvt_f16<<<nb(G_ * I_), 256, 0, stream>>>(eWih, wihE, G_ * I_);
  cvt_f16<<<nb(G_ * H_), 256, 0, stream>>>(eWhh, whhE, G_ * H_);
  cvt_f16<<<nb(G_ * H_), 256, 0, stream>>>(dWhh, whhD, G_ * H_);
  addcvt_f16<<<nb(G_ * H_), 256, 0, stream>>>(dWih, dWhh, wsum, G_ * H_);
  cvt_f16<<<nb(I_ * H_), 256, 0, stream>>>(fcW, fcWh, I_ * H_);
  add_f32<<<nb(G_), 256, 0, stream>>>(ebih, ebhh, bsE, G_);
  add_f32<<<nb(G_), 256, 0, stream>>>(dbih, dbhh, bsD, G_);
  zero16<<<nb(B_ * H_), 256, 0, stream>>>(hE0, B_ * H_);
  zero32<<<nb(B_ * H_), 256, 0, stream>>>(cE, B_ * H_);

  // ---- encoder scan (double-buffered h; c updated in place, tile-exclusive)
  dim3 egrid(B_ / 16, H_ / 16);
  for (int t = 0; t < S_; ++t) {
    _Float16* hin  = (t & 1) ? hE1 : hE0;
    _Float16* hout = (t & 1) ? hE0 : hE1;
    enc_step<<<egrid, 32, 0, stream>>>(xh, t, wihE, whhE, bsE, hin, hout, cE);
  }
  _Float16* hEnc = ((S_ - 1) & 1) ? hE0 : hE1;  // S even -> hE0

  // ---- decoder scan; fc for frame t-1 rides along in step t
  dim3 dgrid(B_ / 16, 10);
  for (int t = 0; t < S_; ++t) {
    const _Float16* A  = (t == 0) ? hEnc : (((t - 1) & 1) ? hD1 : hD0);
    const _Float16* W  = (t == 0) ? whhD : wsum;  // inp==0 at t=0; inp==h after
    _Float16* hout     = (t & 1) ? hD1 : hD0;
    dec_step<<<dgrid, 32, 0, stream>>>(A, W, bsD, cE, hout, A, fcWh, fcb, out, t - 1);
  }
  _Float16* hLast = ((S_ - 1) & 1) ? hD1 : hD0;  // S even -> hD1
  fc_out_kernel<<<dim3(B_ / 16, I_ / 16), 32, 0, stream>>>(hLast, fcWh, fcb, out, S_ - 1);
}
